// DCRNNLayer_15960098472303
// MI455X (gfx1250) — matrix operand order repaired and checked by reading, not compiled, not run on hardware
//
#include <hip/hip_runtime.h>
#include <math.h>

// Problem constants (match reference)
#define BATCH  4
#define NNODES 20000
#define NEDGES 640000
#define INC    32
#define OUTC   64
#define FDIM   96     // INC + OUTC
#define KHOPS  3
#define BN     (BATCH * NNODES)   // 80000, divisible by 16

typedef __attribute__((ext_vector_type(16))) _Float16 v16h;
typedef __attribute__((ext_vector_type(8)))  float    v8f;

// ---------------------------------------------------------------------------
// Build concatenated input: out[b,n,0:32] = x, out[b,n,32:96] = h (optionally
// gated by z, phase 2: r*h with r==z).
// ---------------------------------------------------------------------------
__global__ void concat_gate_kernel(const float* __restrict__ x,
                                   const float* __restrict__ h,
                                   const float* __restrict__ z,   // nullptr in phase 1
                                   float* __restrict__ out) {
    long tid = (long)blockIdx.x * blockDim.x + threadIdx.x;
    if (tid >= (long)BN * FDIM) return;
    int  f    = (int)(tid % FDIM);
    long node = tid / FDIM;                 // flattened b*N + n
    float v;
    if (f < INC) {
        v = x[node * INC + f];
    } else {
        float hv = h[node * OUTC + (f - INC)];
        if (z) hv *= z[node * OUTC + (f - INC)];
        v = hv;
    }
    out[node * FDIM + f] = v;
}

// ---------------------------------------------------------------------------
// One wave32 per (batch, edge). Lane l handles features {l, l+32, l+64}:
// Hnext[b, dst, f] += ew * Hcur[b, src, f]   (L2-resident atomic f32 adds)
// ---------------------------------------------------------------------------
__global__ void diffuse_scatter_kernel(const float* __restrict__ Hcur,
                                       const int*   __restrict__ src,
                                       const int*   __restrict__ dst,
                                       const float* __restrict__ ew,
                                       float*       __restrict__ Hnext) {
    long tid  = (long)blockIdx.x * blockDim.x + threadIdx.x;
    long widx = tid >> 5;                   // wave index == (b, e)
    int  lane = threadIdx.x & 31;
    if (widx >= (long)BATCH * NEDGES) return;
    int e = (int)(widx % NEDGES);
    int b = (int)(widx / NEDGES);
    int s = src[e];
    int d = dst[e];
    float w = ew[e];
    const float* __restrict__ hrow = Hcur + ((long)b * NNODES + s) * FDIM;
    float*       __restrict__ orow = Hnext + ((long)b * NNODES + d) * FDIM;
#pragma unroll
    for (int i = 0; i < 3; ++i) {
        int f = lane + 32 * i;
        atomicAdd(&orow[f], w * hrow[f]);
    }
}

// ---------------------------------------------------------------------------
// acc[BN,64] += Hk[BN,96] @ Wk[96,64] using v_wmma_f32_16x16x32_f16.
// One wave per 16-row tile; 3 K-chunks x 4 column tiles = 12 WMMAs/wave.
// Weights staged per block into LDS *column-major* ([col][K], f16) so each
// lane's 16-half B fragment is one contiguous, 32B-aligned run -> 2x
// ds_load_b128 instead of 16x ds_load_u16.
// ---------------------------------------------------------------------------
__global__ void __launch_bounds__(256)
gemm_acc_wmma_kernel(const float* __restrict__ Hk,
                     const float* __restrict__ Wk,    // weight[k], row-major [96][64]
                     float*       __restrict__ acc) {
    __shared__ _Float16 wlds[OUTC * FDIM];            // [col][K], 6144 halves = 12 KB

    // Stage transposed: wlds[col*96 + K] = Wk[K*64 + col]  (global reads coalesced)
    for (int i = threadIdx.x; i < FDIM * OUTC; i += blockDim.x) {
        int K   = i >> 6;          // / OUTC
        int col = i & (OUTC - 1);
        wlds[col * FDIM + K] = (_Float16)Wk[i];
    }
    __syncthreads();

    int  waveInBlock = threadIdx.x >> 5;
    int  lane        = threadIdx.x & 31;
    long tile        = (long)blockIdx.x * 8 + waveInBlock;
    if (tile >= BN / 16) return;                      // BN/16 = 5000, exact fit

    long rowBase = tile * 16;
    int  mrow    = lane & 15;                         // row (A) / col (B,C) in tile
    int  hiHalf  = lane >> 4;                         // 0: low-K half, 1: high-K half
    const float* __restrict__ hrow = Hk + (rowBase + mrow) * FDIM;

    v8f c[4] = {v8f{}, v8f{}, v8f{}, v8f{}};

#pragma unroll
    for (int kc = 0; kc < FDIM; kc += 32) {
        // A fragment: lane<16 -> K = kc+0..15; lane>=16 -> kc+16..31
        v16h a;
        int ka = kc + hiHalf * 16;
#pragma unroll
        for (int i = 0; i < 8; ++i) {
            a[i]     = (_Float16)hrow[ka + i];
            a[8 + i] = (_Float16)hrow[ka + 8 + i];
        }
        // B fragments: lane<16 -> K = kc+0..15; lane>=16 -> kc+16..31
        // wlds[col*96 + kb .. +15] is contiguous and 32B-aligned.
        int kb = kc + hiHalf * 16;
#pragma unroll
        for (int ct = 0; ct < 4; ++ct) {
            int col = ct * 16 + mrow;
            v16h bf = *(const v16h*)&wlds[col * FDIM + kb];
            c[ct] = __builtin_amdgcn_wmma_f32_16x16x32_f16(
                        /*neg_a=*/false, a, /*neg_b=*/false, bf,
                        /*c_mod=*/(short)0, c[ct],
                        /*reuse_a=*/false, /*reuse_b=*/false);
        }
    }

    // C/D layout: VGPR j -> row = rowBase + j + 8*hiHalf, col = ct*16 + (lane&15)
#pragma unroll
    for (int ct = 0; ct < 4; ++ct) {
        int col = ct * 16 + mrow;
#pragma unroll
        for (int j = 0; j < 8; ++j) {
            long row = rowBase + j + hiHalf * 8;
            acc[row * OUTC + col] += c[ct][j];
        }
    }
}

// ---------------------------------------------------------------------------
// z = sigmoid(acc + bias)
// ---------------------------------------------------------------------------
__global__ void sigmoid_bias_kernel(const float* __restrict__ acc,
                                    const float* __restrict__ bias,
                                    float* __restrict__ z) {
    long tid = (long)blockIdx.x * blockDim.x + threadIdx.x;
    if (tid >= (long)BN * OUTC) return;
    float v = acc[tid] + bias[tid & (OUTC - 1)];
    z[tid] = 1.0f / (1.0f + expf(-v));
}

// ---------------------------------------------------------------------------
// out = z*h + (1-z)*tanh(acc2 + bias)
// ---------------------------------------------------------------------------
__global__ void final_gate_kernel(const float* __restrict__ acc2,
                                  const float* __restrict__ bias,
                                  const float* __restrict__ z,
                                  const float* __restrict__ h,
                                  float* __restrict__ out) {
    long tid = (long)blockIdx.x * blockDim.x + threadIdx.x;
    if (tid >= (long)BN * OUTC) return;
    float ht = tanhf(acc2[tid] + bias[tid & (OUTC - 1)]);
    float zz = z[tid];
    out[tid] = zz * h[tid] + (1.0f - zz) * ht;
}

// ---------------------------------------------------------------------------
extern "C" void kernel_launch(void* const* d_in, const int* in_sizes, int n_in,
                              void* d_out, int out_size, void* d_ws, size_t ws_size,
                              hipStream_t stream) {
    (void)in_sizes; (void)n_in; (void)out_size; (void)ws_size;

    const float* x          = (const float*)d_in[0];   // [B,N,32]
    const float* h          = (const float*)d_in[1];   // [B,N,64]
    const float* weight     = (const float*)d_in[2];   // [3,96,64]
    const float* bias       = (const float*)d_in[3];   // [64]
    const int*   edge_index = (const int*)  d_in[4];   // [2,E]
    const float* ew         = (const float*)d_in[5];   // [E]
    const int*   src = edge_index;
    const int*   dst = edge_index + NEDGES;
    float* out = (float*)d_out;                        // [B,N,64]

    // Workspace layout (floats): xh | ping | pong | acc | z  (~127 MB)
    float* ws     = (float*)d_ws;
    float* W_xh   = ws;
    float* W_ping = W_xh   + (long)BN * FDIM;
    float* W_pong = W_ping + (long)BN * FDIM;
    float* W_acc  = W_pong + (long)BN * FDIM;
    float* W_z    = W_acc  + (long)BN * OUTC;

    const dim3 blk(256);
    const long nCat  = (long)BN * FDIM;
    const int  gCat  = (int)((nCat + 255) / 256);
    const long nOut  = (long)BN * OUTC;
    const int  gOut  = (int)((nOut + 255) / 256);
    const long nScat = (long)BATCH * NEDGES * 32;      // one wave per (b,e)
    const int  gScat = (int)((nScat + 255) / 256);
    const int  gGemm = (BN / 16 + 7) / 8;              // 8 waves per block

    for (int phase = 0; phase < 2; ++phase) {
        const float* zgate = (phase == 0) ? nullptr : W_z;
        concat_gate_kernel<<<gCat, blk, 0, stream>>>(x, h, zgate, W_xh);
        hipMemsetAsync(W_acc, 0, nOut * sizeof(float), stream);

        const float* Hcur  = W_xh;
        float*       Hnext = W_ping;
        for (int k = 0; k < KHOPS; ++k) {
            hipMemsetAsync(Hnext, 0, nCat * sizeof(float), stream);
            diffuse_scatter_kernel<<<gScat, blk, 0, stream>>>(Hcur, src, dst, ew, Hnext);
            gemm_acc_wmma_kernel<<<gGemm, blk, 0, stream>>>(
                Hnext, weight + (long)k * FDIM * OUTC, W_acc);
            Hcur  = Hnext;
            Hnext = (Hnext == W_ping) ? W_pong : W_ping;
        }

        if (phase == 0)
            sigmoid_bias_kernel<<<gOut, blk, 0, stream>>>(W_acc, bias, W_z);
        else
            final_gate_kernel<<<gOut, blk, 0, stream>>>(W_acc, bias, W_z, h, out);
    }
}
